// Attention_549755814415
// MI455X (gfx1250) — matrix operand order repaired and checked by reading, not compiled, not run on hardware
//
#include <hip/hip_runtime.h>
#include <hip/hip_bf16.h>

// ---- problem constants ----
#define CB    8
#define CF    16
#define CNS   196
#define CJ    24
#define CH    8
#define CD    64
#define CDIM  512
#define CN    3160              // CJ + CF*CNS
#define CBH   64                // CB*CH
#define CMT   25280             // CB*CN
#define CTH   1536              // 3*CH*CD
#define CL    220               // CJ + CNS  (spatial kv length)

typedef __attribute__((ext_vector_type(16))) __bf16          v16bf;
typedef __attribute__((ext_vector_type(8)))  float           v8f;
typedef __attribute__((ext_vector_type(16))) unsigned short  v16u;
typedef __attribute__((ext_vector_type(8)))  unsigned short  v8u;
typedef __attribute__((ext_vector_type(4)))  unsigned int    u32x4;
typedef __attribute__((ext_vector_type(4)))  int             i32x4;
typedef __attribute__((ext_vector_type(8)))  int             i32x8;

__device__ __forceinline__ unsigned short f2bf(float f) {
    unsigned u = __builtin_bit_cast(unsigned, f);
    unsigned r = u + 0x7FFFu + ((u >> 16) & 1u);
    return (unsigned short)(r >> 16);
}
__device__ __forceinline__ float bf2f(unsigned short h) {
    return __builtin_bit_cast(float, ((unsigned)h) << 16);
}
__device__ __forceinline__ v8f wmma_bf16(v16u a, v16u b, v8f c) {
    return __builtin_amdgcn_wmma_f32_16x16x32_bf16(
        false, __builtin_bit_cast(v16bf, a),
        false, __builtin_bit_cast(v16bf, b),
        (short)0, c, false, false);
}

// =====================================================================
// Kernel 1: QKV projection  C[25280x1536] = x[25280x512] @ w[512x1536]
// scatter into q/k/v bf16 buffers laid out [BH][N][64]; q scaled 1/8.
// block = 256 thr (8 waves: 4 M-subtiles x 2 N-subtiles), tile 64x128.
// (A needs fp32->bf16 conversion while staging, so no TDM here.)
// =====================================================================
__global__ void qkv_gemm_kernel(const float* __restrict__ x,
                                const float* __restrict__ w,
                                unsigned short* __restrict__ qB,
                                unsigned short* __restrict__ kB,
                                unsigned short* __restrict__ vB) {
    __shared__ __attribute__((aligned(32))) unsigned short sA[64 * 32];
    __shared__ __attribute__((aligned(32))) unsigned short sBt[128 * 32]; // [n][k]

    const int t    = threadIdx.x;
    const int wid  = t >> 5;
    const int lane = t & 31;
    const int wm   = wid & 3;          // M subtile 0..3
    const int wn   = wid >> 2;         // N subtile 0..1
    const int m0   = blockIdx.y * 64;
    const int n0   = blockIdx.x * 128;
    const int hv   = lane >> 4;        // half-wave
    const int lmv  = lane & 15;

    v8f acc[4] = {};

    for (int k0 = 0; k0 < CDIM; k0 += 32) {
        __syncthreads();
        { // stage A 64x32 (bf16-convert x)
            int m  = t >> 2;
            int kk = (t & 3) * 8;
            const float* src = x + (size_t)(m0 + m) * CDIM + k0 + kk;
            unsigned short* dst = sA + m * 32 + kk;
#pragma unroll
            for (int i = 0; i < 8; ++i) dst[i] = f2bf(src[i]);
            if (k0 + 32 < CDIM) __builtin_prefetch(src + 32, 0, 1);
        }
        { // stage B 32x128, transposed into LDS
            int k  = t >> 3;
            int nn = (t & 7) * 16;
            const float* src = w + (size_t)(k0 + k) * CTH + n0 + nn;
#pragma unroll
            for (int i = 0; i < 16; ++i) sBt[(nn + i) * 32 + k] = f2bf(src[i]);
            if (k0 + 32 < CDIM) __builtin_prefetch(src + 32 * CTH, 0, 1);
        }
        __syncthreads();

        v16u a;
        {
            const unsigned short* p = sA + (wm * 16 + lmv) * 32;
            v8u lo = *(const v8u*)(p + hv * 16);
            v8u hi = *(const v8u*)(p + hv * 16 + 8);
#pragma unroll
            for (int i = 0; i < 8; ++i) { a[i] = lo[i]; a[8 + i] = hi[i]; }
        }
#pragma unroll
        for (int j = 0; j < 4; ++j) {
            int n = wn * 64 + j * 16 + lmv;
            v16u b = *(const v16u*)(sBt + n * 32 + hv * 16);
            acc[j] = wmma_bf16(a, b, acc[j]);
        }
    }

    // epilogue: scatter into q/k/v head-major buffers (div hoisted per row)
#pragma unroll
    for (int r = 0; r < 8; ++r) {
        int grow = m0 + wm * 16 + hv * 8 + r;
        int bb   = grow / CN;
        int tok  = grow - bb * CN;
#pragma unroll
        for (int j = 0; j < 4; ++j) {
            int gcol = n0 + wn * 64 + j * 16 + lmv;
            int mat  = gcol >> 9;           // /512
            int c    = gcol & 511;
            int hd   = c >> 6;
            int dc   = c & 63;
            float vv = acc[j][r];
            if (mat == 0) vv *= 0.125f;     // d^-0.5, d=64
            unsigned short* dst = (mat == 0) ? qB : (mat == 1) ? kB : vB;
            dst[(size_t)((bb * CH + hd) * CN + tok) * CD + dc] = f2bf(vv);
        }
    }
}

// =====================================================================
// Kernel 2: joint attention (pre-RoPE k/v): per bh, 24 q rows vs 3160 keys
// block = 64 (2 waves, one 16-row M-tile each), flash over 32-key blocks.
// =====================================================================
__global__ void joint_attn_kernel(const unsigned short* __restrict__ Qb,
                                  const unsigned short* __restrict__ Kb,
                                  const unsigned short* __restrict__ Vb,
                                  unsigned short* __restrict__ attn) {
    __shared__ __attribute__((aligned(32))) unsigned short sP[2][16 * 32];

    const int wid  = threadIdx.x >> 5;
    const int lane = threadIdx.x & 31;
    const int bh   = blockIdx.x;
    const int bb   = bh >> 3, hd = bh & 7;
    const int m0   = wid * 16;
    const int hv   = lane >> 4;
    const int lmv  = lane & 15;

    // q A-fragments (rows m0+lmv; rows >=24 are padding -> zero)
    const int  qr = m0 + lmv;
    const bool qv = (qr < CJ);
    v16u aq0 = {}, aq1 = {};
    if (qv) {
        const unsigned short* qrow = Qb + (size_t)(bh * CN + qr) * CD;
        v8u l0 = *(const v8u*)(qrow + hv * 16);
        v8u h0 = *(const v8u*)(qrow + 8 + hv * 16);
        v8u l1 = *(const v8u*)(qrow + 32 + hv * 16);
        v8u h1 = *(const v8u*)(qrow + 40 + hv * 16);
#pragma unroll
        for (int i = 0; i < 8; ++i) {
            aq0[i] = l0[i]; aq0[8 + i] = h0[i];
            aq1[i] = l1[i]; aq1[8 + i] = h1[i];
        }
    }

    float mrow[8], lrow[8];
#pragma unroll
    for (int r = 0; r < 8; ++r) { mrow[r] = -1e30f; lrow[r] = 0.0f; }
    v8f acc[4] = {};

    for (int kbase = 0; kbase < CN; kbase += 32) {
        v8f s0 = {}, s1 = {};
        { // S tile 0: cols kbase + lmv
            int col = kbase + lmv;
            v16u b0 = {}, b1 = {};
            if (col < CN) {
                const unsigned short* kr = Kb + (size_t)(bh * CN + col) * CD;
                b0 = *(const v16u*)(kr + hv * 16);
                b1 = *(const v16u*)(kr + 32 + hv * 16);
            }
            s0 = wmma_bf16(aq0, b0, s0);
            s0 = wmma_bf16(aq1, b1, s0);
            if (col >= CN) {
#pragma unroll
                for (int r = 0; r < 8; ++r) s0[r] = -1e30f;
            }
        }
        { // S tile 1: cols kbase + 16 + lmv
            int col = kbase + 16 + lmv;
            v16u b0 = {}, b1 = {};
            if (col < CN) {
                const unsigned short* kr = Kb + (size_t)(bh * CN + col) * CD;
                b0 = *(const v16u*)(kr + hv * 16);
                b1 = *(const v16u*)(kr + 32 + hv * 16);
            }
            s1 = wmma_bf16(aq0, b0, s1);
            s1 = wmma_bf16(aq1, b1, s1);
            if (col >= CN) {
#pragma unroll
                for (int r = 0; r < 8; ++r) s1[r] = -1e30f;
            }
        }
        // online softmax per row (row = r + hv*8; 16 lanes per row)
#pragma unroll
        for (int r = 0; r < 8; ++r) {
            float mx = fmaxf(s0[r], s1[r]);
            mx = fmaxf(mx, __shfl_xor(mx, 1));
            mx = fmaxf(mx, __shfl_xor(mx, 2));
            mx = fmaxf(mx, __shfl_xor(mx, 4));
            mx = fmaxf(mx, __shfl_xor(mx, 8));
            float nm = fmaxf(mrow[r], mx);
            float al = __expf(mrow[r] - nm);
            float p0 = __expf(s0[r] - nm);
            float p1 = __expf(s1[r] - nm);
            float rs = p0 + p1;
            rs += __shfl_xor(rs, 1);
            rs += __shfl_xor(rs, 2);
            rs += __shfl_xor(rs, 4);
            rs += __shfl_xor(rs, 8);
            lrow[r] = lrow[r] * al + rs;
            mrow[r] = nm;
            acc[0][r] *= al; acc[1][r] *= al; acc[2][r] *= al; acc[3][r] *= al;
            int prow = r + hv * 8;
            sP[wid][prow * 32 + lmv]      = f2bf(p0);
            sP[wid][prow * 32 + 16 + lmv] = f2bf(p1);
        }
        // re-swizzle P (C-layout) -> A-layout via LDS (per-wave, HW in-order DS)
        v16u ap;
        {
            const unsigned short* pp = &sP[wid][lmv * 32];
            v8u lo = *(const v8u*)(pp + hv * 16);
            v8u hi = *(const v8u*)(pp + hv * 16 + 8);
#pragma unroll
            for (int i = 0; i < 8; ++i) { ap[i] = lo[i]; ap[8 + i] = hi[i]; }
        }
        // P @ V
#pragma unroll
        for (int j = 0; j < 4; ++j) {
            v16u bv = {};
#pragma unroll
            for (int e = 0; e < 16; ++e) {
                int col = kbase + hv * 16 + e;
                if (col < CN)
                    bv[e] = Vb[(size_t)(bh * CN + col) * CD + j * 16 + lmv];
            }
            acc[j] = wmma_bf16(ap, bv, acc[j]);
        }
    }

#pragma unroll
    for (int r = 0; r < 8; ++r) {
        int row = m0 + hv * 8 + r;
        if (row < CJ) {
            float inv = 1.0f / lrow[r];
#pragma unroll
            for (int j = 0; j < 4; ++j) {
                int dim = j * 16 + lmv;
                attn[(size_t)(bb * CN + row) * CDIM + hd * CD + dim] =
                    f2bf(acc[j][r] * inv);
            }
        }
    }
}

// =====================================================================
// Kernel 3: RoPE on spatial q/k (first 32 dims), in place, bf16 buffers
// =====================================================================
__global__ void rope_kernel(unsigned short* __restrict__ qB,
                            unsigned short* __restrict__ kB,
                            const float* __restrict__ sinb,
                            const float* __restrict__ cosb) {
    int tid = blockIdx.x * blockDim.x + threadIdx.x;     // CBH*3136*16 pairs
    if (tid >= CBH * (CF * CNS) * 16) return;
    int p   = tid & 15;
    int row = tid >> 4;
    int bh  = row / (CF * CNS);
    int sp  = row - bh * (CF * CNS);
    int ns  = sp % CNS;
    int tok = CJ + sp;
    size_t base = (size_t)(bh * CN + tok) * CD + p * 2;
    float c0 = cosb[ns * 32 + p * 2], c1 = cosb[ns * 32 + p * 2 + 1];
    float s0 = sinb[ns * 32 + p * 2], s1 = sinb[ns * 32 + p * 2 + 1];

    float x0 = bf2f(qB[base]), x1 = bf2f(qB[base + 1]);
    qB[base]     = f2bf(x0 * c0 - x1 * s0);
    qB[base + 1] = f2bf(x1 * c1 + x0 * s1);

    x0 = bf2f(kB[base]); x1 = bf2f(kB[base + 1]);
    kB[base]     = f2bf(x0 * c0 - x1 * s0);
    kB[base + 1] = f2bf(x1 * c1 + x0 * s1);
}

// =====================================================================
// Kernel 4: spatial attention. 1024 (bh,f) units x 13 M-tiles, 220 keys
// (24 joint + 196 roped spatial). One wave per block.
// =====================================================================
__global__ void spatial_attn_kernel(const unsigned short* __restrict__ Qb,
                                    const unsigned short* __restrict__ Kb,
                                    const unsigned short* __restrict__ Vb,
                                    unsigned short* __restrict__ attn) {
    __shared__ __attribute__((aligned(32))) unsigned short sP[16 * 32];

    const int bid = blockIdx.x;
    const int bhf = bid / 13;
    const int mt  = bid - bhf * 13;
    const int bh  = bhf >> 4;
    const int f   = bhf & 15;
    const int bb  = bh >> 3, hd = bh & 7;
    const int lane = threadIdx.x;
    const int hv = lane >> 4, lmv = lane & 15;
    const int m0 = mt * 16;

    const int  qr = m0 + lmv;
    const bool qv = (qr < CNS);
    v16u aq0 = {}, aq1 = {};
    if (qv) {
        int qtok = CJ + f * CNS + qr;
        const unsigned short* qrow = Qb + (size_t)(bh * CN + qtok) * CD;
        v8u l0 = *(const v8u*)(qrow + hv * 16);
        v8u h0 = *(const v8u*)(qrow + 8 + hv * 16);
        v8u l1 = *(const v8u*)(qrow + 32 + hv * 16);
        v8u h1 = *(const v8u*)(qrow + 40 + hv * 16);
#pragma unroll
        for (int i = 0; i < 8; ++i) {
            aq0[i] = l0[i]; aq0[8 + i] = h0[i];
            aq1[i] = l1[i]; aq1[8 + i] = h1[i];
        }
    }

    float mrow[8], lrow[8];
#pragma unroll
    for (int r = 0; r < 8; ++r) { mrow[r] = -1e30f; lrow[r] = 0.0f; }
    v8f acc[4] = {};

    for (int kbase = 0; kbase < CL; kbase += 32) {
        v8f s0 = {}, s1 = {};
        {
            int col = kbase + lmv;
            v16u b0 = {}, b1 = {};
            if (col < CL) {
                int g = (col < CJ) ? (bh * CN + col)
                                   : (bh * CN + CJ + f * CNS + (col - CJ));
                const unsigned short* kr = Kb + (size_t)g * CD;
                b0 = *(const v16u*)(kr + hv * 16);
                b1 = *(const v16u*)(kr + 32 + hv * 16);
            }
            s0 = wmma_bf16(aq0, b0, s0);
            s0 = wmma_bf16(aq1, b1, s0);
            if (col >= CL) {
#pragma unroll
                for (int r = 0; r < 8; ++r) s0[r] = -1e30f;
            }
        }
        {
            int col = kbase + 16 + lmv;
            v16u b0 = {}, b1 = {};
            if (col < CL) {
                int g = (col < CJ) ? (bh * CN + col)
                                   : (bh * CN + CJ + f * CNS + (col - CJ));
                const unsigned short* kr = Kb + (size_t)g * CD;
                b0 = *(const v16u*)(kr + hv * 16);
                b1 = *(const v16u*)(kr + 32 + hv * 16);
            }
            s1 = wmma_bf16(aq0, b0, s1);
            s1 = wmma_bf16(aq1, b1, s1);
            if (col >= CL) {
#pragma unroll
                for (int r = 0; r < 8; ++r) s1[r] = -1e30f;
            }
        }
#pragma unroll
        for (int r = 0; r < 8; ++r) {
            float mx = fmaxf(s0[r], s1[r]);
            mx = fmaxf(mx, __shfl_xor(mx, 1));
            mx = fmaxf(mx, __shfl_xor(mx, 2));
            mx = fmaxf(mx, __shfl_xor(mx, 4));
            mx = fmaxf(mx, __shfl_xor(mx, 8));
            float nm = fmaxf(mrow[r], mx);
            float al = __expf(mrow[r] - nm);
            float p0 = __expf(s0[r] - nm);
            float p1 = __expf(s1[r] - nm);
            float rs = p0 + p1;
            rs += __shfl_xor(rs, 1);
            rs += __shfl_xor(rs, 2);
            rs += __shfl_xor(rs, 4);
            rs += __shfl_xor(rs, 8);
            lrow[r] = lrow[r] * al + rs;
            mrow[r] = nm;
            acc[0][r] *= al; acc[1][r] *= al; acc[2][r] *= al; acc[3][r] *= al;
            int prow = r + hv * 8;
            sP[prow * 32 + lmv]      = f2bf(p0);
            sP[prow * 32 + 16 + lmv] = f2bf(p1);
        }
        v16u ap;
        {
            const unsigned short* pp = &sP[lmv * 32];
            v8u lo = *(const v8u*)(pp + hv * 16);
            v8u hi = *(const v8u*)(pp + hv * 16 + 8);
#pragma unroll
            for (int i = 0; i < 8; ++i) { ap[i] = lo[i]; ap[8 + i] = hi[i]; }
        }
#pragma unroll
        for (int j = 0; j < 4; ++j) {
            v16u bv = {};
#pragma unroll
            for (int e = 0; e < 16; ++e) {
                int col = kbase + hv * 16 + e;
                if (col < CL) {
                    int g = (col < CJ) ? (bh * CN + col)
                                       : (bh * CN + CJ + f * CNS + (col - CJ));
                    bv[e] = Vb[(size_t)g * CD + j * 16 + lmv];
                }
            }
            acc[j] = wmma_bf16(ap, bv, acc[j]);
        }
    }

#pragma unroll
    for (int r = 0; r < 8; ++r) {
        int row = m0 + hv * 8 + r;
        if (row < CNS) {
            int tok = CJ + f * CNS + row;
            float inv = 1.0f / lrow[r];
#pragma unroll
            for (int j = 0; j < 4; ++j) {
                int dim = j * 16 + lmv;
                attn[(size_t)(bb * CN + tok) * CDIM + hd * CD + dim] =
                    f2bf(acc[j][r] * inv);
            }
        }
    }
}

// =====================================================================
// Kernel 5: output projection  out[25280x512] = attn(bf16) @ w_out + b
// A tile (64x32 bf16, row-major compacted) staged by the Tensor Data
// Mover: wave 0 issues TENSOR_LOAD_TO_LDS (D# per ISA 08 sec 8.3-8.6),
// overlapping with the manual (convert+transpose) B staging, then
// s_wait_tensorcnt + barrier publishes the tile to all 8 waves.
// =====================================================================
__global__ void out_gemm_kernel(const unsigned short* __restrict__ A,
                                const float* __restrict__ w,
                                const float* __restrict__ bias,
                                float* __restrict__ out) {
    __shared__ __attribute__((aligned(32))) unsigned short sA[64 * 32];
    __shared__ __attribute__((aligned(32))) unsigned short sBt[128 * 32];

    const int t    = threadIdx.x;
    const int wid  = t >> 5;
    const int lane = t & 31;
    const int wm   = wid & 3;
    const int wn   = wid >> 2;
    const int m0   = blockIdx.y * 64;
    const int n0   = blockIdx.x * 128;
    const int hv   = lane >> 4;
    const int lmv  = lane & 15;

    // Constant parts of the TDM descriptor (2D bf16 tensor 25280x512,
    // tile 32(dim0=K) x 64(dim1=M), dense row-major into LDS).
    const unsigned lds_addr = (unsigned)(size_t)(&sA[0]);
    i32x8 g1;
    g1[0] = (int)(1u << 16);                         // data_size=1 (2 bytes)
    g1[1] = (int)((unsigned)CDIM << 16);             // tensor_dim0[15:0]
    g1[2] = (int)(((unsigned)CDIM >> 16) | ((unsigned)CMT << 16)); // dim0 hi | dim1 lo
    g1[3] = (int)(((unsigned)CMT >> 16) | (32u << 16));            // dim1 hi | tile_dim0=32
    g1[4] = (int)(64u);                              // tile_dim1=64 (tile_dim2=0)
    g1[5] = (int)((unsigned)CDIM);                   // tensor_dim0_stride low32
    g1[6] = 0;
    g1[7] = 0;
    const i32x4 gz4 = {};                            // groups 2/3: 2D tensor
    const i32x8 gz8 = {};                            // extra group (clang-23 form)

    v8f acc[4] = {};

    for (int k0 = 0; k0 < CDIM; k0 += 32) {
        __syncthreads();                 // previous tile fully consumed
        if (wid == 0) {                  // TDM: async A-tile -> LDS
            unsigned long long ga = (unsigned long long)(size_t)A +
                ((size_t)m0 * CDIM + (size_t)k0) * 2ull;
            u32x4 g0;
            g0[0] = 1u;                                  // count=1 (valid D#)
            g0[1] = lds_addr;
            g0[2] = (unsigned)(ga & 0xFFFFFFFFu);
            g0[3] = (unsigned)((ga >> 32) & 0x01FFFFFFu) | (2u << 30); // type=2
            __builtin_amdgcn_tensor_load_to_lds(g0, g1, gz4, gz4, gz8, 0);
        }
        { // stage B 32x128 fp32 -> bf16, transposed into LDS (all waves)
            int k  = t >> 3;
            int nn = (t & 7) * 16;
            const float* src = w + (size_t)(k0 + k) * CDIM + n0 + nn;
#pragma unroll
            for (int i = 0; i < 16; ++i) sBt[(nn + i) * 32 + k] = f2bf(src[i]);
            if (k0 + 32 < CDIM) __builtin_prefetch(src + 32 * CDIM, 0, 1);
        }
        if (wid == 0) __builtin_amdgcn_s_wait_tensorcnt(0);
        __syncthreads();                 // publish A tile + B tile

        v16u a;
        {
            const unsigned short* p = sA + (wm * 16 + lmv) * 32;
            v8u lo = *(const v8u*)(p + hv * 16);
            v8u hi = *(const v8u*)(p + hv * 16 + 8);
#pragma unroll
            for (int i = 0; i < 8; ++i) { a[i] = lo[i]; a[8 + i] = hi[i]; }
        }
#pragma unroll
        for (int j = 0; j < 4; ++j) {
            int n = wn * 64 + j * 16 + lmv;
            v16u b = *(const v16u*)(sBt + n * 32 + hv * 16);
            acc[j] = wmma_bf16(a, b, acc[j]);
        }
    }

#pragma unroll
    for (int j = 0; j < 4; ++j) {
#pragma unroll
        for (int r = 0; r < 8; ++r) {
            int grow = m0 + wm * 16 + hv * 8 + r;
            int gcol = n0 + wn * 64 + j * 16 + lmv;
            out[(size_t)grow * CDIM + gcol] = acc[j][r] + bias[gcol];
        }
    }
}

// =====================================================================
extern "C" void kernel_launch(void* const* d_in, const int* in_sizes, int n_in,
                              void* d_out, int out_size, void* d_ws, size_t ws_size,
                              hipStream_t stream) {
    (void)in_sizes; (void)n_in; (void)out_size; (void)ws_size;
    const float* x     = (const float*)d_in[0];
    const float* w_qkv = (const float*)d_in[1];
    const float* w_out = (const float*)d_in[2];
    const float* b_out = (const float*)d_in[3];
    const float* sinb  = (const float*)d_in[4];
    const float* cosb  = (const float*)d_in[5];
    float* out = (float*)d_out;

    // workspace: q, k, v [BH][N][64] bf16 + attn [B][N][512] bf16 (~104 MB)
    const size_t nq = (size_t)CBH * CN * CD;        // 12,943,360 elems each
    unsigned short* qB    = (unsigned short*)d_ws;
    unsigned short* kB    = qB + nq;
    unsigned short* vB    = kB + nq;
    unsigned short* attnB = vB + nq;

    // 1. QKV projection (25280x512 @ 512x1536)
    qkv_gemm_kernel<<<dim3(CTH / 128, CMT / 64), 256, 0, stream>>>(
        x, w_qkv, qB, kB, vB);

    // 2. joint attention (must precede RoPE: uses pre-RoPE k/v)
    joint_attn_kernel<<<CBH, 64, 0, stream>>>(qB, kB, vB, attnB);

    // 3. RoPE on spatial q/k
    {
        int total = CBH * (CF * CNS) * 16;          // 3,211,264
        rope_kernel<<<(total + 255) / 256, 256, 0, stream>>>(qB, kB, sinb, cosb);
    }

    // 4. spatial attention: 1024 units x 13 M-tiles
    spatial_attn_kernel<<<1024 * 13, 32, 0, stream>>>(qB, kB, vB, attnB);

    // 5. output projection + bias
    out_gemm_kernel<<<dim3(CDIM / 128, CMT / 64), 256, 0, stream>>>(
        attnB, w_out, b_out, out);
}
